// MAB_74663711473764
// MI455X (gfx1250) — compile-verified, matrix operand order checked by reading, not run on hardware
//
#include <hip/hip_runtime.h>
#include <math.h>

// MAB (multihead attention block) for MI455X / gfx1250, wave32 + WMMA + TDM.
// B=32, N=512, D=512, H=8, dh=64. All heavy math via v_wmma_f32_16x16x32_f16;
// GEMM tile staging via tensor_load_to_lds (Tensor Data Mover), double-buffered.

typedef _Float16 v16h __attribute__((ext_vector_type(16)));
typedef float    v8f  __attribute__((ext_vector_type(8)));
typedef unsigned int u32x4 __attribute__((ext_vector_type(4)));
typedef int          i32x4 __attribute__((ext_vector_type(4)));
typedef int          i32x8 __attribute__((ext_vector_type(8)));

static constexpr float kScale = 0.044194173824159216f; // 1/sqrt(512)

__device__ __forceinline__ v8f wmma_f16(v16h a, v16h b, v8f c) {
  // (neg_a, A, neg_b, B, c_mod, C, reuse_a, reuse_b)
  return __builtin_amdgcn_wmma_f32_16x16x32_f16(false, a, false, b, (short)0, c,
                                                false, false);
}

// 16-bit A/B fragment for 16x16x32 WMMA. Per ISA layout, lane (lo=L%16,
// hi=L/16) holds K=[8*hi,8*hi+8) in VGPR0-3 and K=[16+8*hi,+8) in VGPR4-7.
__device__ __forceinline__ v16h ldfrag(const _Float16* rowp, int hi) {
  v16h a;
  *reinterpret_cast<uint4*>(&a)       = *reinterpret_cast<const uint4*>(rowp + 8 * hi);
  *(reinterpret_cast<uint4*>(&a) + 1) = *reinterpret_cast<const uint4*>(rowp + 16 + 8 * hi);
  return a;
}

// ---------------------------------------------------------------------------
// TDM: DMA a 128-row x 32-col f16 tile (row stride 512 elements in memory)
// into LDS, padding each 64B row with 16B -> LDS row stride = 40 halves.
// D# built per CDNA5 ISA 8.3-8.6; issued by one wave, tracked by TENSORcnt.
// This toolchain exposes the 6-arg builtin: (g0, g1, g2, g3, extra, cpol).
// ---------------------------------------------------------------------------
__device__ __forceinline__ void tdm_load_tile(const _Float16* gsrc,
                                              unsigned lds_byte_off) {
  const unsigned long long ga = (unsigned long long)(uintptr_t)gsrc;
  u32x4 g0;
  g0.x = 1u;                                   // count=1 (valid user desc)
  g0.y = lds_byte_off;                         // lds_addr
  g0.z = (unsigned)ga;                         // global_addr[31:0]
  g0.w = (unsigned)(ga >> 32) | (2u << 30);    // global_addr[56:32] | type=2
  i32x8 g1;
  g1[0] = (1 << 16)      // data_size = 2 bytes
        | (1 << 20)      // pad_enable
        | (3 << 22)      // pad_interval: every 16 DWORDs (64B = one tile row)
        | (3 << 25);     // pad_amount: 4 DWORDs (16B -> 40-half LDS stride)
  g1[1] = (512 << 16);                 // tensor_dim0 = 512 (low 16 in [31:16])
  g1[2] = 0;                           // tensor_dim0 hi | tensor_dim1 lo
  g1[3] = 0x10 | (32 << 16);           // tensor_dim1 = 1<<20 (hi16) | tile_dim0=32
  g1[4] = 128;                         // tile_dim1 = 128, tile_dim2 = 0
  g1[5] = 512;                         // tensor_dim0_stride = 512 elements
  g1[6] = 0;
  g1[7] = 0;
  const i32x4 z4 = {0, 0, 0, 0};               // groups 2/3 unused (2D tile)
  const i32x8 z8 = {0, 0, 0, 0, 0, 0, 0, 0};   // extra group slot unused
  __builtin_amdgcn_tensor_load_to_lds(g0, g1, z4, z4, z8, 0);
}

// ---------------------------------------------------------------------------
// f32 -> f16 elementwise convert (8 elems/thread).
// ---------------------------------------------------------------------------
__global__ __launch_bounds__(256) void cvt16(const float* __restrict__ s,
                                             _Float16* __restrict__ d, int n) {
  const int i = (blockIdx.x * 256 + threadIdx.x) * 8;
  if (i >= n) return;
  const float4 a = reinterpret_cast<const float4*>(s + i)[0];
  const float4 b = reinterpret_cast<const float4*>(s + i)[1];
  _Float16 t[8] = {(_Float16)a.x, (_Float16)a.y, (_Float16)a.z, (_Float16)a.w,
                   (_Float16)b.x, (_Float16)b.y, (_Float16)b.z, (_Float16)b.w};
  *reinterpret_cast<uint4*>(d + i) = *reinterpret_cast<uint4*>(t);
}

// ---------------------------------------------------------------------------
// GEMM: C[M,N] = epi( A[M,K] @ W[N,K]^T + bias[N] ) (+ R residual), all-f16
// operands, TDM double-buffered tile staging, 128x128 C tile, K-step 32.
//   EPI: 0 bias, 1 bias+ReLU, 2 bias+residual R (f32)
//   OH : 0 f32 out, 1 f16 out, 2 dual (f32 -> Cp, f16 -> C2p)
// 8 waves in 2x4 grid; each wave: 64x32 via 8 f32 accumulators.
// ---------------------------------------------------------------------------
template <int EPI, int OH>
__global__ __launch_bounds__(256) void gemm_f16(const _Float16* __restrict__ A16,
                                                const _Float16* __restrict__ W16,
                                                const float* __restrict__ bias,
                                                const float* __restrict__ R,
                                                void* __restrict__ Cp,
                                                void* __restrict__ C2p,
                                                int M, int N, int K) {
  __shared__ _Float16 As[2][128][40];
  __shared__ _Float16 Ws[2][128][40];

  const int t = threadIdx.x;
  const int lane = t & 31;
  const int wave = t >> 5;
  const int lo = lane & 15;
  const int hi = lane >> 4;
  const int wm = (wave >> 2) * 64;
  const int wn = (wave & 3) * 32;
  const int m0 = blockIdx.y * 128;
  const int n0 = blockIdx.x * 128;

  const _Float16* Abase = A16 + (size_t)m0 * K;
  const _Float16* Wbase = W16 + (size_t)n0 * K;
  const unsigned offA0 = (unsigned)(uintptr_t)&As[0][0][0];
  const unsigned offA1 = (unsigned)(uintptr_t)&As[1][0][0];
  const unsigned offW0 = (unsigned)(uintptr_t)&Ws[0][0][0];
  const unsigned offW1 = (unsigned)(uintptr_t)&Ws[1][0][0];

  const int NK = K >> 5;  // # of 32-wide k tiles

  if (wave == 0) {  // prologue: DMA tile 0 into buffer 0
    tdm_load_tile(Abase, offA0);
    tdm_load_tile(Wbase, offW0);
  }

  v8f acc[4][2] = {};

  for (int kt = 0; kt < NK; ++kt) {
    const int cur = kt & 1;
    __syncthreads();  // all waves done reading buffer cur^1
    if (wave == 0) {
      if (kt + 1 < NK) {  // prefetch next tile into the other buffer
        tdm_load_tile(Abase + (kt + 1) * 32, cur ? offA0 : offA1);
        tdm_load_tile(Wbase + (kt + 1) * 32, cur ? offW0 : offW1);
        __builtin_amdgcn_s_wait_tensorcnt(2);  // tiles for 'cur' landed
      } else {
        __builtin_amdgcn_s_wait_tensorcnt(0);
      }
    }
    __syncthreads();  // buffer 'cur' visible to all waves

    v16h af[4], wf[2];
#pragma unroll
    for (int i = 0; i < 4; ++i) af[i] = ldfrag(&As[cur][wm + 16 * i + lo][0], hi);
#pragma unroll
    for (int j = 0; j < 2; ++j) wf[j] = ldfrag(&Ws[cur][wn + 16 * j + lo][0], hi);
#pragma unroll
    for (int i = 0; i < 4; ++i)
#pragma unroll
      for (int j = 0; j < 2; ++j) acc[i][j] = wmma_f16(af[i], wf[j], acc[i][j]);
  }

  // epilogue: C layout m = r + 8*hi, n = lo
#pragma unroll
  for (int j = 0; j < 2; ++j) {
    const int gc = n0 + wn + 16 * j + lo;
    const float bv = bias[gc];
#pragma unroll
    for (int i = 0; i < 4; ++i) {
      const int gr = m0 + wm + 16 * i + 8 * hi;
#pragma unroll
      for (int r = 0; r < 8; ++r) {
        float x = acc[i][j][r] + bv;
        if constexpr (EPI == 1) x = fmaxf(x, 0.0f);
        const size_t idx = (size_t)(gr + r) * N + gc;
        if constexpr (EPI == 2) x += R[idx];
        if constexpr (OH == 0) {
          ((float*)Cp)[idx] = x;
        } else if constexpr (OH == 1) {
          ((_Float16*)Cp)[idx] = (_Float16)x;
        } else {
          ((float*)Cp)[idx] = x;
          ((_Float16*)C2p)[idx] = (_Float16)x;
        }
      }
    }
  }
}

// ---------------------------------------------------------------------------
// Phase 1: per-(b,h,k-column) softmax stats over the QUERY axis.
// One wave owns 16 k-columns; k-fragments resident, stream 32 q-tiles through
// WMMA, online max/sum with cross-half shfl_xor(16).
// ---------------------------------------------------------------------------
__global__ __launch_bounds__(256) void attn_stats(const _Float16* __restrict__ q2,
                                                  const _Float16* __restrict__ k2,
                                                  float* __restrict__ Mst,
                                                  float* __restrict__ Lst) {
  const int blk = blockIdx.x;  // bh*4 + kb
  const int bh = blk >> 2;
  const int kb = blk & 3;
  const int b = bh >> 3;
  const int h = bh & 7;
  const int t = threadIdx.x;
  const int lane = t & 31;
  const int wave = t >> 5;
  const int lo = lane & 15;
  const int hi = lane >> 4;

  const int col = kb * 128 + wave * 16 + lo;
  const _Float16* krow = k2 + (((size_t)b * 512 + col) * 512) + h * 64;
  const v16h bk0 = ldfrag(krow, hi);
  const v16h bk1 = ldfrag(krow + 32, hi);

  float m_run = -1e30f, l_run = 0.0f;

  for (int q0 = 0; q0 < 512; q0 += 16) {
    const _Float16* qrow = q2 + (((size_t)b * 512 + q0 + lo) * 512) + h * 64;
    const v16h aq0 = ldfrag(qrow, hi);
    const v16h aq1 = ldfrag(qrow + 32, hi);
    v8f s = {};
    s = wmma_f16(aq0, bk0, s);
    s = wmma_f16(aq1, bk1, s);

    float pmax = -1e30f;
#pragma unroll
    for (int r = 0; r < 8; ++r) pmax = fmaxf(pmax, s[r] * kScale);
    pmax = fmaxf(pmax, __shfl_xor(pmax, 16, 32));
    const float nm = fmaxf(m_run, pmax);
    float ps = 0.0f;
#pragma unroll
    for (int r = 0; r < 8; ++r) ps += __expf(s[r] * kScale - nm);
    ps += __shfl_xor(ps, 16, 32);
    l_run = l_run * __expf(m_run - nm) + ps;
    m_run = nm;
  }
  if (lane < 16) {
    Mst[(size_t)bh * 512 + col] = m_run;
    Lst[(size_t)bh * 512 + col] = l_run;
  }
}

// ---------------------------------------------------------------------------
// Phase 2: attn = P @ V with P = exp(scale*S - M[k]) / L[k], S recomputed by
// WMMA; P round-trips wave-private LDS (C->A layout), V is LDS-transposed.
// ---------------------------------------------------------------------------
__global__ __launch_bounds__(256) void attn_av(const _Float16* __restrict__ q2,
                                               const _Float16* __restrict__ k2,
                                               const _Float16* __restrict__ v2,
                                               const float* __restrict__ Mst,
                                               const float* __restrict__ Lst,
                                               _Float16* __restrict__ at2) {
  __shared__ _Float16 vT[64][40];
  __shared__ _Float16 Pw[8][16][40];

  const int blk = blockIdx.x;  // bh*8 + qb
  const int bh = blk >> 3;
  const int qb = blk & 7;
  const int b = bh >> 3;
  const int h = bh & 7;
  const int t = threadIdx.x;
  const int lane = t & 31;
  const int wave = t >> 5;
  const int lo = lane & 15;
  const int hi = lane >> 4;

  const int q0  = qb * 64 + (wave & 3) * 16;
  const int dh0 = (wave >> 2) * 32;

  const _Float16* qrow = q2 + (((size_t)b * 512 + q0 + lo) * 512) + h * 64;
  const v16h aq0 = ldfrag(qrow, hi);
  const v16h aq1 = ldfrag(qrow + 32, hi);

  v8f oacc[2] = {};

  const int vtok = t >> 3;
  const int vd0  = (t & 7) * 8;

  for (int kt = 0; kt < 512; kt += 32) {
    __syncthreads();
    {
      const _Float16* vsrc =
          v2 + (((size_t)b * 512 + kt + vtok) * 512) + h * 64 + vd0;
      _Float16 tmp[8];
      *reinterpret_cast<uint4*>(tmp) = *reinterpret_cast<const uint4*>(vsrc);
#pragma unroll
      for (int e = 0; e < 8; ++e) vT[vd0 + e][vtok] = tmp[e];
    }
    __syncthreads();

#pragma unroll
    for (int jn = 0; jn < 2; ++jn) {
      const int ktok = kt + jn * 16 + lo;
      const _Float16* krow = k2 + (((size_t)b * 512 + ktok) * 512) + h * 64;
      const v16h bk0 = ldfrag(krow, hi);
      const v16h bk1 = ldfrag(krow + 32, hi);
      v8f s = {};
      s = wmma_f16(aq0, bk0, s);
      s = wmma_f16(aq1, bk1, s);
      const float Mv = Mst[(size_t)bh * 512 + ktok];
      const float Li = 1.0f / Lst[(size_t)bh * 512 + ktok];
#pragma unroll
      for (int r = 0; r < 8; ++r)
        Pw[wave][r + 8 * hi][jn * 16 + lo] =
            (_Float16)(__expf(s[r] * kScale - Mv) * Li);
    }

    const v16h ap = ldfrag(&Pw[wave][lo][0], hi);
#pragma unroll
    for (int jd = 0; jd < 2; ++jd) {
      const v16h bv = ldfrag(&vT[dh0 + jd * 16 + lo][0], hi);
      oacc[jd] = wmma_f16(ap, bv, oacc[jd]);
    }
  }

#pragma unroll
  for (int jd = 0; jd < 2; ++jd) {
#pragma unroll
    for (int r = 0; r < 8; ++r) {
      const int qrw = q0 + 8 * hi + r;
      const int dcl = h * 64 + dh0 + jd * 16 + lo;
      at2[(((size_t)b * 512 + qrw) * 512) + dcl] = (_Float16)oacc[jd][r];
    }
  }
}

// ---------------------------------------------------------------------------
extern "C" void kernel_launch(void* const* d_in, const int* in_sizes, int n_in,
                              void* d_out, int out_size, void* d_ws,
                              size_t ws_size, hipStream_t stream) {
  (void)in_sizes; (void)n_in; (void)out_size; (void)ws_size;
  const float* Q  = (const float*)d_in[0];
  const float* Kx = (const float*)d_in[1];
  const float* Wq = (const float*)d_in[2];
  const float* bq = (const float*)d_in[3];
  const float* Wk = (const float*)d_in[4];
  const float* bk = (const float*)d_in[5];
  const float* Wv = (const float*)d_in[6];
  const float* bv = (const float*)d_in[7];
  const float* Wo = (const float*)d_in[8];
  const float* bo = (const float*)d_in[9];
  const float* W1 = (const float*)d_in[10];
  const float* b1 = (const float*)d_in[11];
  const float* W2 = (const float*)d_in[12];
  const float* b2 = (const float*)d_in[13];

  constexpr int B = 32, Nt = 512, D = 512, H = 8;
  constexpr int M = B * Nt;                          // 16384
  const size_t szH = (size_t)M * D * sizeof(_Float16);  // activations, f16
  const size_t szW = (size_t)D * D * sizeof(_Float16);  // weights, f16
  const size_t szF = (size_t)M * D * sizeof(float);

  char* p = (char*)d_ws;
  _Float16* Q16  = (_Float16*)p; p += szH;
  _Float16* K16  = (_Float16*)p; p += szH;
  _Float16* q2   = (_Float16*)p; p += szH;
  _Float16* k2   = (_Float16*)p; p += szH;
  _Float16* v2   = (_Float16*)p; p += szH;
  _Float16* at2  = (_Float16*)p; p += szH;
  _Float16* outh = (_Float16*)p; p += szH;
  _Float16* h1   = (_Float16*)p; p += szH;
  _Float16* Wq6  = (_Float16*)p; p += szW;
  _Float16* Wk6  = (_Float16*)p; p += szW;
  _Float16* Wv6  = (_Float16*)p; p += szW;
  _Float16* Wo6  = (_Float16*)p; p += szW;
  _Float16* W16w = (_Float16*)p; p += szW;
  _Float16* W26w = (_Float16*)p; p += szW;
  float*    outb = (float*)p;    p += szF;
  float*    Mst  = (float*)p;    p += (size_t)B * H * Nt * sizeof(float);
  float*    Lst  = (float*)p;

  const dim3 blk(256);
  const dim3 gg(D / 128, M / 128);  // (4, 128)
  const int nAct = M * D;           // 8388608
  const int nWgt = D * D;           // 262144

  // One-time f32 -> f16 conversions (activations + weights).
  cvt16<<<dim3(nAct / 2048), blk, 0, stream>>>(Q,  Q16, nAct);
  cvt16<<<dim3(nAct / 2048), blk, 0, stream>>>(Kx, K16, nAct);
  cvt16<<<dim3(nWgt / 2048), blk, 0, stream>>>(Wq, Wq6, nWgt);
  cvt16<<<dim3(nWgt / 2048), blk, 0, stream>>>(Wk, Wk6, nWgt);
  cvt16<<<dim3(nWgt / 2048), blk, 0, stream>>>(Wv, Wv6, nWgt);
  cvt16<<<dim3(nWgt / 2048), blk, 0, stream>>>(Wo, Wo6, nWgt);
  cvt16<<<dim3(nWgt / 2048), blk, 0, stream>>>(W1, W16w, nWgt);
  cvt16<<<dim3(nWgt / 2048), blk, 0, stream>>>(W2, W26w, nWgt);

  // q/k/v projections (TDM-staged f16 GEMMs, f16 out)
  gemm_f16<0, 1><<<gg, blk, 0, stream>>>(Q16, Wq6, bq, nullptr, q2, nullptr, M, D, D);
  gemm_f16<0, 1><<<gg, blk, 0, stream>>>(K16, Wk6, bk, nullptr, k2, nullptr, M, D, D);
  gemm_f16<0, 1><<<gg, blk, 0, stream>>>(K16, Wv6, bv, nullptr, v2, nullptr, M, D, D);

  // attention: column softmax stats (query axis), then P @ V
  attn_stats<<<dim3(B * H * 4), blk, 0, stream>>>(q2, k2, Mst, Lst);
  attn_av<<<dim3(B * H * 8), blk, 0, stream>>>(q2, k2, v2, Mst, Lst, at2);

  // out = Q + attn @ Wo^T + bo  -> f32 (residual source) + f16 (FFN input)
  gemm_f16<2, 2><<<gg, blk, 0, stream>>>(at2, Wo6, bo, Q, outb, outh, M, D, D);
  // h1 = relu(out @ W1^T + b1) -> f16
  gemm_f16<1, 1><<<gg, blk, 0, stream>>>(outh, W16w, b1, nullptr, h1, nullptr, M, D, D);
  // final = out + h1 @ W2^T + b2 -> f32 d_out
  gemm_f16<2, 0><<<gg, blk, 0, stream>>>(h1, W26w, b2, outb, (float*)d_out, nullptr, M, D, D);
}
